// TwoSimplicialAttention_33019708571649
// MI455X (gfx1250) — matrix operand order repaired and checked by reading, not compiled, not run on hardware
//
#include <hip/hip_runtime.h>
#include <hip/hip_bf16.h>

// ---------------------------------------------------------------------------
// 2-simplicial attention block for gfx1250 (MI455X), bf16 WMMA pipeline.
// N=4096, IN=OUT=1024, H=16, D=64, W1=W2=16.
// ---------------------------------------------------------------------------

#define NROWS   4096
#define DIM     1024
#define NHEADS  16
#define HDIM    64
#define WIN     16

typedef __attribute__((ext_vector_type(16))) __bf16 v16bf;
typedef __attribute__((ext_vector_type(8)))  __bf16 v8bf;
typedef __attribute__((ext_vector_type(4)))  __bf16 v4bf;
typedef __attribute__((ext_vector_type(8)))  float  v8f;

static __device__ __forceinline__ __bf16 f2bf(float f) { return (__bf16)f; }
static __device__ __forceinline__ v8bf  ld8(const __bf16* p) { return *(const v8bf*)p; }
static __device__ __forceinline__ void  st8(__bf16* p, v8bf v) { *(v8bf*)p = v; }

// ---------------------------------------------------------------------------
// Async global->LDS copy (CDNA5 ASYNCcnt path) with compile-time fallbacks.
// Builtin signature (from hipcc diagnostic): param0 = int4* in global AS,
// param1 = int4* in LDS AS, then immediate offset + cpol.
// ---------------------------------------------------------------------------
#if defined(__HIP_DEVICE_COMPILE__) && defined(__has_builtin)
#  if __has_builtin(__builtin_amdgcn_global_load_async_to_lds_b128)
#    define ASYNC_MODE 2
#  elif __has_builtin(__builtin_amdgcn_load_to_lds)
#    define ASYNC_MODE 1
#  endif
#endif
#ifndef ASYNC_MODE
#  define ASYNC_MODE 0
#endif

typedef int v4i_vec __attribute__((vector_size(4 * sizeof(int))));
typedef __attribute__((address_space(1))) v4i_vec g_v4i;
typedef __attribute__((address_space(3))) v4i_vec l_v4i;

static __device__ __forceinline__ void cp128(__bf16* lds, const __bf16* g) {
#if ASYNC_MODE == 2
    __builtin_amdgcn_global_load_async_to_lds_b128(
        (g_v4i*)const_cast<__bf16*>(g), (l_v4i*)lds, 0, 0);
#elif ASYNC_MODE == 1
    __builtin_amdgcn_load_to_lds((g_v4i*)const_cast<__bf16*>(g),
                                 (l_v4i*)lds, 16, 0, 0);
#else
    *(v8bf*)lds = *(const v8bf*)g;
#endif
}

static __device__ __forceinline__ void cp_wait() {
#if defined(__HIP_DEVICE_COMPILE__) && defined(__has_builtin)
#  if __has_builtin(__builtin_amdgcn_s_wait_asynccnt)
    __builtin_amdgcn_s_wait_asynccnt(0);
#  endif
#endif
}

// ---------------------------------------------------------------------------
// f32 -> bf16 conversion (vectorized: float4 in, 4x bf16 out)
// ---------------------------------------------------------------------------
__global__ void cvt_f32_bf16(const float4* __restrict__ src, v4bf* __restrict__ dst, int n4) {
    int i = blockIdx.x * blockDim.x + threadIdx.x;
    if (i < n4) {
        float4 f = src[i];
        v4bf o;
        o[0] = f2bf(f.x); o[1] = f2bf(f.y); o[2] = f2bf(f.z); o[3] = f2bf(f.w);
        dst[i] = o;
    }
}

// ---------------------------------------------------------------------------
// f32 [k][n] -> bf16 transposed [n][k]; 32x32 LDS tile, block (32,8).
// Makes the GEMM's B-fragment K-contiguous so all staging is b128 copies.
// ---------------------------------------------------------------------------
__global__ void cvt_transpose_bf16(const float* __restrict__ W, __bf16* __restrict__ Wt) {
    __shared__ __bf16 t[32][33];
    const int k0 = blockIdx.x * 32;
    const int n0 = blockIdx.y * 32;
    const int tx = threadIdx.x;   // 0..31
    const int ty = threadIdx.y;   // 0..7
    for (int i = ty; i < 32; i += 8)
        t[i][tx] = f2bf(W[(size_t)(k0 + i) * DIM + n0 + tx]);
    __syncthreads();
    for (int i = ty; i < 32; i += 8)
        Wt[(size_t)(n0 + i) * DIM + k0 + tx] = t[tx][i];
}

// ---------------------------------------------------------------------------
// GEMM: C[4096,1024] = A[4096,1024] @ W[1024,1024] (+bias) (+resid)
// W passed PRE-TRANSPOSED (Wt[n][k]).
// MODE 0: bf16 output (projections).  MODE 1: f32 output = acc+bias+resid.
// Block: 128 threads (4 waves), tile 64x64, K-step 32, double-buffered LDS
// filled by async global->LDS b128 copies (ASYNCcnt) when available.
// ---------------------------------------------------------------------------
template <int MODE>
__global__ void gemm_bf16_k(const __bf16* __restrict__ A,
                            const __bf16* __restrict__ Wt,
                            const float*  __restrict__ bias,
                            const float*  __restrict__ resid,
                            __bf16* __restrict__ outb,
                            float*  __restrict__ outf) {
    __shared__ __bf16 As[2][64][32];   // A tile, row-major [m][k]
    __shared__ __bf16 Bs[2][64][32];   // W tile, [n][k] (from Wt rows)

    const int tid  = threadIdx.x;       // 0..127
    const int wave = tid >> 5;          // 0..3
    const int lane = tid & 31;
    const int m0   = blockIdx.x * 64;
    const int n0   = blockIdx.y * 64;

    v8f acc[4] = {};

    const int m   = wave * 16 + (lane & 15);    // A-fragment row
    const int akb = (lane < 16) ? 0 : 8;        // A K-split base
    const int bkb = (lane < 16) ? 0 : 16;       // unused: B loads use akb
    const int nc  = lane & 15;                  // B/D column within 16

    // stage one 64x32 x2 tile pair: 512 b128 copies, 4 per thread
    auto stage = [&](int buf, int kk) {
        for (int t = tid; t < 256; t += 128) {
            int r = t >> 2, c8 = (t & 3) * 8;
            cp128(&As[buf][r][c8], A  + (size_t)(m0 + r) * DIM + kk + c8);
            cp128(&Bs[buf][r][c8], Wt + (size_t)(n0 + r) * DIM + kk + c8);
        }
    };

    stage(0, 0);
    const int NSTEP = DIM / 32;
    for (int step = 0; step < NSTEP; ++step) {
        cp_wait();
        __syncthreads();                       // tile `step` staged & visible
        if (step + 1 < NSTEP)
            stage((step + 1) & 1, (step + 1) * 32);   // fill other buffer async
        const int buf = step & 1;

        // A fragment: two contiguous 8-element runs (ds_load_b128)
        v8bf a0 = ld8(&As[buf][m][akb]);
        v8bf a1 = ld8(&As[buf][m][16 + akb]);
        v16bf a;
#pragma unroll
        for (int e = 0; e < 8; ++e) { a[e] = a0[e]; a[8 + e] = a1[e]; }

#pragma unroll
        for (int cch = 0; cch < 4; ++cch) {
            v8bf b0 = ld8(&Bs[buf][cch * 16 + nc][akb]);
            v8bf b1 = ld8(&Bs[buf][cch * 16 + nc][16 + akb]);
            v16bf b;
#pragma unroll
            for (int e = 0; e < 8; ++e) { b[e] = b0[e]; b[8 + e] = b1[e]; }
            acc[cch] = __builtin_amdgcn_wmma_f32_16x16x32_bf16(
                false, a, false, b, (short)0, acc[cch], false, false);
        }
    }

    // epilogue: D layout — lane holds col nc, rows r + 8*(lane>=16)
    const int hi = (lane >= 16) ? 8 : 0;
#pragma unroll
    for (int cch = 0; cch < 4; ++cch) {
        const int col = n0 + cch * 16 + nc;
        const float bcol = bias[col];
#pragma unroll
        for (int r = 0; r < 8; ++r) {
            int row = m0 + wave * 16 + r + hi;
            float v = acc[cch][r] + bcol;
            if (MODE == 0) {
                outb[(size_t)row * DIM + col] = f2bf(v);
            } else {
                outf[(size_t)row * DIM + col] = v + resid[(size_t)row * DIM + col];
            }
        }
    }
}

// ---------------------------------------------------------------------------
// Attention body: one wave per (n, h).
//   U[j][d] = Q[n][d]*K[n-15+j][d]   -> A = U(16x64) @ Kp_win(64x16): 2 WMMA
//   masked 2D softmax (shfl_xor, wave32)
//   T = S(16x16) @ Vp_win(16x64): 4 WMMA (K zero-padded to 32)
//   Z[d] = sum_j V_j[d]*T[j][d]: VALU + shfl_xor(16)
// FULL=true: n >= 15, no padding anywhere (hot path, no masks).
// ---------------------------------------------------------------------------
template <bool FULL>
__device__ __forceinline__ void attn_body(
        const __bf16* __restrict__ Q,  const __bf16* __restrict__ K,
        const __bf16* __restrict__ V,  const __bf16* __restrict__ Kp,
        const __bf16* __restrict__ Vp, __bf16* __restrict__ Z,
        int n, int h, int lane,
        __bf16 (*sS)[16], __bf16 (*sV)[64], __bf16 (*sVp)[64]) {
    const int  hb  = h * HDIM;
    const int  jk  = lane & 15;          // row j (A-frag) / col k (B-frag) / col (D)
    const bool hi  = lane >= 16;
    const int  akb = hi ? 8 : 0;
    const int  bkb = hi ? 16 : 0;

    const int  rowj   = n - (WIN - 1) + jk;
    const bool jvalid = FULL || (rowj >= 0);
    const int  rj     = FULL ? rowj : (rowj < 0 ? 0 : rowj);

    const __bf16* qrow  = Q  + (size_t)n  * DIM + hb;
    const __bf16* krow  = K  + (size_t)rj * DIM + hb;
    const __bf16* kprow = Kp + (size_t)rj * DIM + hb;

    // ---- stage V / Vp windows into LDS (b128 loads) ----
    for (int t = lane; t < 128; t += 32) {
        int r = t >> 3, c8 = (t & 7) * 8;
        int rv = n - (WIN - 1) + r;
        v8bf vv = {}, pv = {};
        if (FULL || rv >= 0) {
            vv = ld8(V  + (size_t)rv * DIM + hb + c8);
            pv = ld8(Vp + (size_t)rv * DIM + hb + c8);
        }
        st8(&sV[r][c8],  vv);
        st8(&sVp[r][c8], pv);
    }

    // ---- scores A = U @ Kp_win ----
    v8f acc = {};
#pragma unroll
    for (int s = 0; s < 2; ++s) {
        const int ks = s * 32;
        v8bf q0 = ld8(qrow  + ks + akb);
        v8bf q1 = ld8(qrow  + ks + 16 + akb);
        v8bf k0 = ld8(krow  + ks + akb);
        v8bf k1 = ld8(krow  + ks + 16 + akb);
        v8bf p0 = ld8(kprow + ks + akb);
        v8bf p1 = ld8(kprow + ks + 16 + akb);
        if (!jvalid) {
            v8bf z = {};
            k0 = z; k1 = z; p0 = z; p1 = z;
        }
        v16bf a, b;
#pragma unroll
        for (int e = 0; e < 8; ++e) {
            a[e]     = f2bf((float)q0[e] * (float)k0[e]);
            a[8 + e] = f2bf((float)q1[e] * (float)k1[e]);
            b[e]     = p0[e];
            b[8 + e] = p1[e];
        }
        acc = __builtin_amdgcn_wmma_f32_16x16x32_bf16(
            false, a, false, b, (short)0, acc, false, false);
    }

    // ---- mask + 2D softmax over 256 entries ----
    const float NEG = -3.0e38f;
    float vals[8];
    float mx = NEG;
#pragma unroll
    for (int r = 0; r < 8; ++r) {
        float vv;
        if (FULL) {
            vv = acc[r] * 0.125f;                       // 1/sqrt(64)
        } else {
            int j = r + (hi ? 8 : 0);
            bool valid = jvalid && (n - (WIN - 1) + j >= 0);
            vv = valid ? acc[r] * 0.125f : NEG;
        }
        vals[r] = vv;
        mx = fmaxf(mx, vv);
    }
#pragma unroll
    for (int off = 16; off; off >>= 1) mx = fmaxf(mx, __shfl_xor(mx, off, 32));
    float p[8], sum = 0.f;
#pragma unroll
    for (int r = 0; r < 8; ++r) {
        p[r] = FULL ? __expf(vals[r] - mx)
                    : ((vals[r] <= NEG) ? 0.f : __expf(vals[r] - mx));
        sum += p[r];
    }
#pragma unroll
    for (int off = 16; off; off >>= 1) sum += __shfl_xor(sum, off, 32);
    const float inv = 1.f / sum;
#pragma unroll
    for (int r = 0; r < 8; ++r)
        sS[r + (hi ? 8 : 0)][jk] = f2bf(p[r] * inv);    // S[j][k]
    __syncthreads();

    // ---- S as A-fragment (16x16 in a 16x32 A, upper K half zero) ----
    v16bf sa;
#pragma unroll
    for (int e = 0; e < 16; ++e)
        sa[e] = (e < 8) ? sS[jk][akb + e] : f2bf(0.f);

    // ---- T = S @ Vp_win, then Z[d] = sum_j V_j[d]*T[j][d] ----
#pragma unroll
    for (int cch = 0; cch < 4; ++cch) {
        const int dcol = cch * 16 + jk;
        v16bf vb = {};
        {
#pragma unroll
            for (int e = 0; e < 8; ++e)
                vb[e] = sVp[akb + e][dcol];             // K index akb + e, rest zero
        }
        v8f t = {};
        t = __builtin_amdgcn_wmma_f32_16x16x32_bf16(
            false, sa, false, vb, (short)0, t, false, false);

        float part = 0.f;
#pragma unroll
        for (int r = 0; r < 8; ++r)
            part += (float)sV[r + (hi ? 8 : 0)][dcol] * t[r];
        part += __shfl_xor(part, 16, 32);
        if (!hi) Z[(size_t)n * DIM + hb + dcol] = f2bf(part);
    }
}

__global__ void attn_k(const __bf16* __restrict__ Q,  const __bf16* __restrict__ K,
                       const __bf16* __restrict__ V,  const __bf16* __restrict__ Kp,
                       const __bf16* __restrict__ Vp, __bf16* __restrict__ Z) {
    __shared__ __bf16 sS[16][16];
    __shared__ __bf16 sV[16][64];
    __shared__ __bf16 sVp[16][64];
    const int n = blockIdx.x, h = blockIdx.y, lane = threadIdx.x;
    if (n >= WIN - 1)  // uniform branch: 4081 of 4096 blocks take the maskless path
        attn_body<true >(Q, K, V, Kp, Vp, Z, n, h, lane, sS, sV, sVp);
    else
        attn_body<false>(Q, K, V, Kp, Vp, Z, n, h, lane, sS, sV, sVp);
}

// ---------------------------------------------------------------------------
// LayerNorm, in-place on f32 buffer [4096,1024]; one block (256 thr) per row.
// ---------------------------------------------------------------------------
__global__ void ln_k(float* __restrict__ buf, const float* __restrict__ gamma,
                     const float* __restrict__ beta) {
    const int row = blockIdx.x;
    const int tid = threadIdx.x;           // 0..255
    __shared__ float r1[8], r2[8];
    float* p = buf + (size_t)row * DIM;

    float s = 0.f, s2 = 0.f;
    for (int i = tid; i < DIM; i += 256) {
        float v = p[i];
        s += v; s2 += v * v;
    }
#pragma unroll
    for (int off = 16; off; off >>= 1) {
        s  += __shfl_xor(s,  off, 32);
        s2 += __shfl_xor(s2, off, 32);
    }
    if ((tid & 31) == 0) { r1[tid >> 5] = s; r2[tid >> 5] = s2; }
    __syncthreads();
    float ts = 0.f, ts2 = 0.f;
#pragma unroll
    for (int i = 0; i < 8; ++i) { ts += r1[i]; ts2 += r2[i]; }
    const float mean = ts * (1.f / DIM);
    const float var  = ts2 * (1.f / DIM) - mean * mean;
    const float rinv = rsqrtf(var + 1e-5f);
    for (int i = tid; i < DIM; i += 256)
        p[i] = (p[i] - mean) * rinv * gamma[i] + beta[i];
}

// ---------------------------------------------------------------------------
// Host launcher
// ---------------------------------------------------------------------------
extern "C" void kernel_launch(void* const* d_in, const int* in_sizes, int n_in,
                              void* d_out, int out_size, void* d_ws, size_t ws_size,
                              hipStream_t stream) {
    (void)in_sizes; (void)n_in; (void)out_size; (void)ws_size;

    const float* x     = (const float*)d_in[0];
    const float* Wq    = (const float*)d_in[1];
    const float* bq    = (const float*)d_in[2];
    const float* Wk    = (const float*)d_in[3];
    const float* bk    = (const float*)d_in[4];
    const float* Wv    = (const float*)d_in[5];
    const float* bv    = (const float*)d_in[6];
    const float* Wkp   = (const float*)d_in[7];
    const float* bkp   = (const float*)d_in[8];
    const float* Wvp   = (const float*)d_in[9];
    const float* bvp   = (const float*)d_in[10];
    const float* Wo    = (const float*)d_in[11];
    const float* bo    = (const float*)d_in[12];
    const float* gamma = (const float*)d_in[13];
    const float* beta  = (const float*)d_in[14];

    const size_t NX = (size_t)NROWS * DIM;     // 4096*1024
    const size_t NW = (size_t)DIM * DIM;       // 1024*1024

    char* ws = (char*)d_ws;
    size_t off = 0;
    auto bump = [&](size_t bytes) -> void* {
        void* pp = (void*)(ws + off);
        off = (off + bytes + 255) & ~(size_t)255;
        return pp;
    };
    __bf16* xb   = (__bf16*)bump(NX * 2);
    __bf16* wqt  = (__bf16*)bump(NW * 2);
    __bf16* wkt  = (__bf16*)bump(NW * 2);
    __bf16* wvt  = (__bf16*)bump(NW * 2);
    __bf16* wkpt = (__bf16*)bump(NW * 2);
    __bf16* wvpt = (__bf16*)bump(NW * 2);
    __bf16* wot  = (__bf16*)bump(NW * 2);
    __bf16* qb   = (__bf16*)bump(NX * 2);
    __bf16* kb   = (__bf16*)bump(NX * 2);
    __bf16* vb   = (__bf16*)bump(NX * 2);
    __bf16* kpb  = (__bf16*)bump(NX * 2);
    __bf16* vpb  = (__bf16*)bump(NX * 2);
    __bf16* zb   = (__bf16*)bump(NX * 2);

    // 1) convert x to bf16; convert+transpose weights to bf16 [n][k]
    const int nx4 = (int)(NX / 4);
    cvt_f32_bf16<<<(nx4 + 255) / 256, 256, 0, stream>>>((const float4*)x, (v4bf*)xb, nx4);
    dim3 tg(DIM / 32, DIM / 32), tb(32, 8);
    cvt_transpose_bf16<<<tg, tb, 0, stream>>>(Wq,  wqt);
    cvt_transpose_bf16<<<tg, tb, 0, stream>>>(Wk,  wkt);
    cvt_transpose_bf16<<<tg, tb, 0, stream>>>(Wv,  wvt);
    cvt_transpose_bf16<<<tg, tb, 0, stream>>>(Wkp, wkpt);
    cvt_transpose_bf16<<<tg, tb, 0, stream>>>(Wvp, wvpt);
    cvt_transpose_bf16<<<tg, tb, 0, stream>>>(Wo,  wot);

    // 2) five projection GEMMs (WMMA bf16, f32 accumulate, bf16 out)
    dim3 gg(NROWS / 64, DIM / 64);
    gemm_bf16_k<0><<<gg, 128, 0, stream>>>(xb, wqt,  bq,  nullptr, qb,  nullptr);
    gemm_bf16_k<0><<<gg, 128, 0, stream>>>(xb, wkt,  bk,  nullptr, kb,  nullptr);
    gemm_bf16_k<0><<<gg, 128, 0, stream>>>(xb, wvt,  bv,  nullptr, vb,  nullptr);
    gemm_bf16_k<0><<<gg, 128, 0, stream>>>(xb, wkpt, bkp, nullptr, kpb, nullptr);
    gemm_bf16_k<0><<<gg, 128, 0, stream>>>(xb, wvpt, bvp, nullptr, vpb, nullptr);

    // 3) windowed trilinear attention, one wave per (n, h)
    attn_k<<<dim3(NROWS, NHEADS), 32, 0, stream>>>(qb, kb, vb, kpb, vpb, zb);

    // 4) output GEMM + bias + residual -> d_out (f32)
    gemm_bf16_k<1><<<gg, 128, 0, stream>>>(zb, wot, bo, x, nullptr, (float*)d_out);

    // 5) LayerNorm in place on d_out
    ln_k<<<NROWS, 256, 0, stream>>>((float*)d_out, gamma, beta);
}
